// MoE_9517647528570
// MI455X (gfx1250) — matrix operand order repaired and checked by reading, not compiled, not run on hardware
//
#include <hip/hip_runtime.h>
#include <math.h>

// ---------------------------------------------------------------------------
// MoE (top-2 of 8 experts), T=16384 tokens, E=512 model dim, H=768 hidden.
// bf16 WMMA (v_wmma_f32_16x16x32_bf16) for both expert GEMMs, fp32 accum,
// exact GELU, deterministic fp32 atomic combine (exactly 2 adds per element).
// M-tile = 32 tokens: every B fragment from L2 feeds two WMMAs (halves L2
// weight traffic vs a 16-token tile).
// ---------------------------------------------------------------------------

#define TOKENS 16384   // B*N = 8*2048
#define EDIM   512     // model dim
#define HDIM   768     // hidden dim
#define NEXP   8       // experts
#define MT     32      // tokens per FFN tile (two WMMA M tiles)
#define XPAD   8       // LDS row padding (ushorts) for x tile
#define HPAD   8       // LDS row padding (ushorts) for h tile

typedef __attribute__((ext_vector_type(16))) __bf16        v16bf;
typedef __attribute__((ext_vector_type(8)))  float         v8f;
typedef __attribute__((ext_vector_type(4)))  unsigned int  u32x4;

union FragU { u32x4 q[2]; v16bf v; };

__device__ __forceinline__ unsigned short f2bf(float f) {
    unsigned int u = __float_as_uint(f);
    u += 0x7FFFu + ((u >> 16) & 1u);   // round-to-nearest-even
    return (unsigned short)(u >> 16);
}

// ---------------------------------------------------------------------------
// Convert + transpose weights to bf16, WMMA-B-friendly layout (K contiguous
// per output column):  w1t[x][h][e] = W1[x][e][h];  w2t[x][e][h] = W2[x][h][e]
// ---------------------------------------------------------------------------
__global__ void convert_weights(const float* __restrict__ W1,
                                const float* __restrict__ W2,
                                unsigned short* __restrict__ w1t,
                                unsigned short* __restrict__ w2t) {
    const int NW = NEXP * HDIM * EDIM;
    int idx = blockIdx.x * blockDim.x + threadIdx.x;
    if (idx < NW) {
        int x = idx / (HDIM * EDIM);
        int r = idx - x * (HDIM * EDIM);
        int h = r / EDIM;
        int e = r - h * EDIM;
        w1t[idx] = f2bf(W1[(x * EDIM + e) * HDIM + h]);
    } else {
        int j = idx - NW;
        if (j < NW) {
            int x = j / (EDIM * HDIM);
            int r = j - x * (EDIM * HDIM);
            int e = r / HDIM;
            int h = r - e * HDIM;
            w2t[j] = f2bf(W2[(x * HDIM + h) * EDIM + e]);
        }
    }
}

// ---------------------------------------------------------------------------
// Router: one wave32 per token. Scores over 8 experts, top-2 (first-index
// tie-break like jax.lax.top_k), softmax over the 2 logits, bucket append.
// ---------------------------------------------------------------------------
__global__ __launch_bounds__(256)
void router_kernel(const float* __restrict__ x,
                   const float* __restrict__ Wg,
                   const float* __restrict__ bg,
                   int*   __restrict__ counts,
                   int*   __restrict__ tlist,
                   float* __restrict__ glist) {
    const int lane  = threadIdx.x & 31;
    const int wave  = threadIdx.x >> 5;
    const int token = blockIdx.x * 8 + wave;
    if (token >= TOKENS) return;

    float s[NEXP];
#pragma unroll
    for (int j = 0; j < NEXP; ++j) s[j] = 0.f;

    const float* xp = x + (size_t)token * EDIM;
    for (int e = lane; e < EDIM; e += 32) {
        float xv = xp[e];
        const float* wr = Wg + e * NEXP;
#pragma unroll
        for (int j = 0; j < NEXP; ++j) s[j] += xv * wr[j];
    }
#pragma unroll
    for (int off = 16; off >= 1; off >>= 1) {
#pragma unroll
        for (int j = 0; j < NEXP; ++j) s[j] += __shfl_xor(s[j], off, 32);
    }
    if (lane == 0) {
#pragma unroll
        for (int j = 0; j < NEXP; ++j) s[j] += bg[j];
        int i0 = 0; float v0 = s[0];
#pragma unroll
        for (int j = 1; j < NEXP; ++j) if (s[j] > v0) { v0 = s[j]; i0 = j; }
        int i1 = 0; float v1 = -3.402823e38f;
#pragma unroll
        for (int j = 0; j < NEXP; ++j) if (j != i0 && s[j] > v1) { v1 = s[j]; i1 = j; }
        float e1  = expf(v1 - v0);           // v0 >= v1
        float inv = 1.f / (1.f + e1);
        float g0 = inv, g1 = e1 * inv;

        int s0 = atomicAdd(&counts[i0], 1);
        tlist[i0 * TOKENS + s0] = token;  glist[i0 * TOKENS + s0] = g0;
        int s1 = atomicAdd(&counts[i1], 1);
        tlist[i1 * TOKENS + s1] = token;  glist[i1 * TOKENS + s1] = g1;
    }
}

// ---------------------------------------------------------------------------
// Expert FFN: one block = 32 tokens of one expert, 8 waves, bf16 WMMA.
// GEMM1 [32,512]x[512,768] -> GELU -> GEMM2 [32,768]x[768,512] -> gated
// atomic scatter-add into out.  Each B fragment is reused for both M halves.
// ---------------------------------------------------------------------------
__global__ __launch_bounds__(256)
void moe_ffn_kernel(const float* __restrict__ x,
                    const unsigned short* __restrict__ w1t,
                    const unsigned short* __restrict__ w2t,
                    const float* __restrict__ b1,
                    const float* __restrict__ b2,
                    const int*   __restrict__ counts,
                    const int*   __restrict__ tlist,
                    const float* __restrict__ glist,
                    float* __restrict__ out) {
    const int e    = blockIdx.y;
    const int tile = blockIdx.x;
    const int cnt  = counts[e];
    if (tile * MT >= cnt) return;

    __shared__ unsigned short xs[MT * (EDIM + XPAD)];   // 32.5 KB bf16 x tile
    __shared__ unsigned short hs[MT * (HDIM + HPAD)];   // 48.5 KB bf16 h tile
    __shared__ int   tok_s[MT];
    __shared__ float gate_s[MT];

    const int tid  = threadIdx.x;
    const int lane = tid & 31;
    const int wave = tid >> 5;
    const int lh   = lane & 15;    // N column / M row within 16
    const int lg   = lane >> 4;    // lane-group (0: lanes 0-15, 1: lanes 16-31)

    if (tid < MT) {
        int row = tile * MT + tid;
        tok_s[tid]  = (row < cnt) ? tlist[e * TOKENS + row] : 0;
        gate_s[tid] = (row < cnt) ? glist[e * TOKENS + row] : 0.f;   // pad rows: gate 0
    }
    // Gather x tile -> LDS as bf16 (coalesced along model dim).
    for (int i = tid; i < MT * EDIM; i += 256) {
        int m = i >> 9;                   // / EDIM
        int k = i & (EDIM - 1);
        int row = tile * MT + m;
        int token = (row < cnt) ? tlist[e * TOKENS + row] : 0;
        xs[m * (EDIM + XPAD) + k] = f2bf(x[(size_t)token * EDIM + k]);
    }
    __syncthreads();

    // ---------------- GEMM1 + GELU:  h = gelu(x @ W1 + b1) -----------------
    const unsigned short* w1e = w1t + (size_t)e * HDIM * EDIM;  // [h][e]
    for (int nt = wave; nt < HDIM / 16; nt += 8) {
        const int n0 = nt * 16;
        v8f acc0 = {0.f, 0.f, 0.f, 0.f, 0.f, 0.f, 0.f, 0.f};
        v8f acc1 = {0.f, 0.f, 0.f, 0.f, 0.f, 0.f, 0.f, 0.f};
        // A: per ISA 16-bit 16x32 layout -> lane (lh,lg): K chunks lg*8 and lg*8+16
        const unsigned short* a0 = xs + lh        * (EDIM + XPAD) + lg * 8;
        const unsigned short* a1 = xs + (16 + lh) * (EDIM + XPAD) + lg * 8;
        // B: 32x16 layout -> lane owns column n0+lh, K chunks lg*8 and lg*8+16
        const unsigned short* bp = w1e + (size_t)(n0 + lh) * EDIM + lg * 8;
        for (int k0 = 0; k0 < EDIM; k0 += 32) {
            FragU b, a;
            b.q[0] = *(const u32x4*)(bp + k0);        // K = k0+lg*8    .. +7
            b.q[1] = *(const u32x4*)(bp + k0 + 16);   // K = k0+16+lg*8 .. +7
            a.q[0] = *(const u32x4*)(a0 + k0);        // K = k0+lg*8    .. +7
            a.q[1] = *(const u32x4*)(a0 + k0 + 16);   // K = k0+16+lg*8 .. +7
            acc0 = __builtin_amdgcn_wmma_f32_16x16x32_bf16(
                false, a.v, false, b.v, (short)0, acc0, false, false);
            a.q[0] = *(const u32x4*)(a1 + k0);
            a.q[1] = *(const u32x4*)(a1 + k0 + 16);
            acc1 = __builtin_amdgcn_wmma_f32_16x16x32_bf16(
                false, a.v, false, b.v, (short)0, acc1, false, false);
        }
        const float bb = b1[e * HDIM + n0 + lh];
#pragma unroll
        for (int r = 0; r < 8; ++r) {                 // C layout: M = r + lg*8
            float v0 = acc0[r] + bb;
            float g0 = 0.5f * v0 * (1.0f + erff(v0 * 0.70710678118654752f));
            hs[(lg * 8 + r) * (HDIM + HPAD) + n0 + lh] = f2bf(g0);
            float v1 = acc1[r] + bb;
            float g1 = 0.5f * v1 * (1.0f + erff(v1 * 0.70710678118654752f));
            hs[(16 + lg * 8 + r) * (HDIM + HPAD) + n0 + lh] = f2bf(g1);
        }
    }
    __syncthreads();

    // ---------------- GEMM2 + gate + combine:  out += g * (h @ W2 + b2) ----
    const unsigned short* w2e = w2t + (size_t)e * EDIM * HDIM;  // [e][h]
    for (int nt = wave; nt < EDIM / 16; nt += 8) {
        const int n0 = nt * 16;
        v8f acc0 = {0.f, 0.f, 0.f, 0.f, 0.f, 0.f, 0.f, 0.f};
        v8f acc1 = {0.f, 0.f, 0.f, 0.f, 0.f, 0.f, 0.f, 0.f};
        const unsigned short* a0 = hs + lh        * (HDIM + HPAD) + lg * 8;
        const unsigned short* a1 = hs + (16 + lh) * (HDIM + HPAD) + lg * 8;
        const unsigned short* bp = w2e + (size_t)(n0 + lh) * HDIM + lg * 8;
        for (int k0 = 0; k0 < HDIM; k0 += 32) {
            FragU b, a;
            b.q[0] = *(const u32x4*)(bp + k0);
            b.q[1] = *(const u32x4*)(bp + k0 + 16);
            a.q[0] = *(const u32x4*)(a0 + k0);
            a.q[1] = *(const u32x4*)(a0 + k0 + 16);
            acc0 = __builtin_amdgcn_wmma_f32_16x16x32_bf16(
                false, a.v, false, b.v, (short)0, acc0, false, false);
            a.q[0] = *(const u32x4*)(a1 + k0);
            a.q[1] = *(const u32x4*)(a1 + k0 + 16);
            acc1 = __builtin_amdgcn_wmma_f32_16x16x32_bf16(
                false, a.v, false, b.v, (short)0, acc1, false, false);
        }
        const float bb = b2[e * EDIM + n0 + lh];
#pragma unroll
        for (int r = 0; r < 8; ++r) {
            int m = lg * 8 + r;
            float val0 = gate_s[m] * (acc0[r] + bb);        // pad rows: gate 0
            __hip_atomic_fetch_add(&out[(size_t)tok_s[m] * EDIM + n0 + lh], val0,
                                   __ATOMIC_RELAXED, __HIP_MEMORY_SCOPE_AGENT);
            float val1 = gate_s[16 + m] * (acc1[r] + bb);
            __hip_atomic_fetch_add(&out[(size_t)tok_s[16 + m] * EDIM + n0 + lh], val1,
                                   __ATOMIC_RELAXED, __HIP_MEMORY_SCOPE_AGENT);
        }
    }
}

// ---------------------------------------------------------------------------
extern "C" void kernel_launch(void* const* d_in, const int* in_sizes, int n_in,
                              void* d_out, int out_size, void* d_ws, size_t ws_size,
                              hipStream_t stream) {
    (void)in_sizes; (void)n_in; (void)out_size; (void)ws_size;
    const float* x  = (const float*)d_in[0];
    const float* Wg = (const float*)d_in[1];
    const float* bg = (const float*)d_in[2];
    const float* W1 = (const float*)d_in[3];
    const float* b1 = (const float*)d_in[4];
    const float* W2 = (const float*)d_in[5];
    const float* b2 = (const float*)d_in[6];
    float* out = (float*)d_out;

    char* ws = (char*)d_ws;
    size_t off = 0;
    int*   counts = (int*)(ws + off);              off += 256;
    int*   tlist  = (int*)(ws + off);              off += (size_t)NEXP * TOKENS * 4;
    float* glist  = (float*)(ws + off);            off += (size_t)NEXP * TOKENS * 4;
    unsigned short* w1t = (unsigned short*)(ws + off); off += (size_t)NEXP * HDIM * EDIM * 2;
    unsigned short* w2t = (unsigned short*)(ws + off); off += (size_t)NEXP * EDIM * HDIM * 2;

    hipMemsetAsync(counts, 0, 256, stream);
    hipMemsetAsync(out, 0, (size_t)TOKENS * EDIM * sizeof(float), stream);

    const int total = 2 * NEXP * HDIM * EDIM;
    convert_weights<<<(total + 255) / 256, 256, 0, stream>>>(W1, W2, w1t, w2t);

    router_kernel<<<TOKENS / 8, 256, 0, stream>>>(x, Wg, bg, counts, tlist, glist);

    dim3 grid(TOKENS / MT, NEXP);   // blocks with tile*32 >= count[e] exit early
    moe_ffn_kernel<<<grid, 256, 0, stream>>>(x, w1t, w2t, b1, b2,
                                             counts, tlist, glist, out);
}
